// EmbeddingTable_57183194579270
// MI455X (gfx1250) — compile-verified, matrix operand order checked by reading, not run on hardware
//
#include <hip/hip_runtime.h>

// Problem constants (from reference setup_inputs)
#define D_IN   1024
#define D_OUT  4096
#define VOCAB  32000
#define NROWS  4096      // B*S = 2*2048
#define M_BLK  32        // rows per workgroup
#define VT     512       // vocab tile width (32 waves * 16 cols)
#define NTHREADS 1024
#define NWAVES 32

typedef __attribute__((ext_vector_type(16))) __bf16 v16bf;
typedef __attribute__((ext_vector_type(8)))  __bf16 v8bf;
typedef __attribute__((ext_vector_type(8)))  float  v8f;
typedef __attribute__((ext_vector_type(4)))  unsigned v4u;
typedef __attribute__((ext_vector_type(8)))  int      v8i;
typedef __attribute__((ext_vector_type(4)))  int      v4i;

// ---------------- DPP16 row_xmask butterfly reductions ----------------
// XOR-shuffle within each 16-lane DPP row (exactly our reduction domain:
// lanes 0-15 and 16-31 reduce independently). Pure VALU, no LDS port.
template <int MASK>
__device__ __forceinline__ float dpp_xor(float v) {
  int i = __builtin_bit_cast(int, v);
  int p = __builtin_amdgcn_update_dpp(i, i, 0x160 | MASK, 0xF, 0xF, true);
  return __builtin_bit_cast(float, p);
}
__device__ __forceinline__ float row16_max(float v) {
  v = fmaxf(v, dpp_xor<1>(v));
  v = fmaxf(v, dpp_xor<2>(v));
  v = fmaxf(v, dpp_xor<4>(v));
  v = fmaxf(v, dpp_xor<8>(v));
  return v;
}
__device__ __forceinline__ float row16_sum(float v) {
  v += dpp_xor<1>(v);
  v += dpp_xor<2>(v);
  v += dpp_xor<4>(v);
  v += dpp_xor<8>(v);
  return v;
}

// ---------------- fragment loaders ----------------
// A/B fragment from a bf16 row-major buffer (LDS or global). Per-lane element j
// holds k = kb + j (j<8) and k = kb + 16 + (j-8); caller passes p = row_base + kb.
// Both runs are 16B-aligned -> ds_load_b128 / global_load_b128.
__device__ __forceinline__ v16bf frag_bf16(const __bf16* p) {
  v8bf lo = *(const v8bf*)(p);
  v8bf hi = *(const v8bf*)(p + 16);
  v16bf a;
#pragma unroll
  for (int j = 0; j < 8; ++j) { a[j] = lo[j]; a[8 + j] = hi[j]; }
  return a;
}

// ---------------- prep kernels (one-time, memory bound) ----------------
// f32 -> bf16 elementwise (W_ff and x). n must be a multiple of 8.
__global__ void __launch_bounds__(256)
cvt_f32_bf16(const float* __restrict__ src, __bf16* __restrict__ dst, int n) {
  int i = (blockIdx.x * 256 + threadIdx.x) * 8;
  if (i + 8 <= n) {
    float4 f0 = *(const float4*)(src + i);
    float4 f1 = *(const float4*)(src + i + 4);
    v8bf o;
    o[0] = (__bf16)f0.x; o[1] = (__bf16)f0.y; o[2] = (__bf16)f0.z; o[3] = (__bf16)f0.w;
    o[4] = (__bf16)f1.x; o[5] = (__bf16)f1.y; o[6] = (__bf16)f1.z; o[7] = (__bf16)f1.w;
    *(v8bf*)(dst + i) = o;
  }
}

// emb [VOCAB x D_OUT] f32 -> embT [D_OUT x VOCAB] bf16, LDS-tiled 64x64.
#define TT 64
__global__ void __launch_bounds__(256)
transpose_emb(const float* __restrict__ emb, __bf16* __restrict__ embT) {
  __shared__ __bf16 tile[TT][TT + 1];
  const int v0 = blockIdx.x * TT;          // vocab base
  const int n0 = blockIdx.y * TT;          // d_out base
  const int tx = threadIdx.x & 63;
  const int ty = threadIdx.x >> 6;         // 4 row-groups
#pragma unroll
  for (int r = ty; r < TT; r += 4)         // read coalesced along emb rows
    tile[tx][r] = (__bf16)emb[(size_t)(v0 + r) * D_OUT + (n0 + tx)];
  __syncthreads();
#pragma unroll
  for (int r = ty; r < TT; r += 4)         // write coalesced along embT rows
    embT[(size_t)(n0 + r) * VOCAB + (v0 + tx)] = tile[r][tx];
}

// ---------------- fused main kernel ----------------
// Two 16x16 logits tiles (rows 0-15 / 16-31) for this wave's 16 vocab columns.
__device__ __forceinline__ void logits_tile(const __bf16* __restrict__ xt,
                                            const __bf16* __restrict__ Wb,
                                            const float* __restrict__ bff,
                                            int v0, int w, int lane,
                                            v8f& s0, v8f& s1) {
  const int half = lane >> 4;
  const int l16  = lane & 15;
  const int vrow = v0 + w * 16 + l16;
  const int vrc  = (vrow < VOCAB) ? vrow : (VOCAB - 1);
  const __bf16* wrow = Wb + (size_t)vrc * D_IN;
  // prefetch this wave's W row for the next vocab tile (streams through L2)
  {
    int vn = vrow + VT; vn = (vn < VOCAB) ? vn : (VOCAB - 1);
    __builtin_prefetch(Wb + (size_t)vn * D_IN, 0, 1);
  }
#pragma unroll
  for (int e = 0; e < 8; ++e) { s0[e] = 0.0f; s1[e] = 0.0f; }
#pragma unroll 4
  for (int k0 = 0; k0 < D_IN; k0 += 32) {
    const int kb = k0 + half * 8;
    v16bf b  = frag_bf16(wrow + kb);
    v16bf a0 = frag_bf16(xt + (size_t)l16 * D_IN + kb);
    v16bf a1 = frag_bf16(xt + (size_t)(16 + l16) * D_IN + kb);
    s0 = __builtin_amdgcn_wmma_f32_16x16x32_bf16(false, a0, false, b, (short)0, s0, false, false);
    s1 = __builtin_amdgcn_wmma_f32_16x16x32_bf16(false, a1, false, b, (short)0, s1, false, false);
  }
  const float bias = (vrow < VOCAB) ? bff[vrow] : 0.0f;
#pragma unroll
  for (int e = 0; e < 8; ++e) { s0[e] += bias; s1[e] += bias; }
}

// LDS layout (dynamic):  [0,64K)=xt  [64K,96K)=pt  [96K,100K)=red  then stats
#define SM_XT   0
#define SM_PT   (M_BLK * D_IN * 2)                  // 65536
#define SM_RED  (SM_PT + M_BLK * VT * 2)            // 98304
#define SM_STAT (SM_RED + NWAVES * M_BLK * 4)       // 102400
#define SM_TOTAL (SM_STAT + 4 * M_BLK * 4)          // 102912 bytes

__global__ void __launch_bounds__(NTHREADS)
fused_ff_softmax_emb(const __bf16* __restrict__ xb, const __bf16* __restrict__ Wb,
                     const float* __restrict__ bff, const __bf16* __restrict__ embT,
                     float* __restrict__ out) {
  extern __shared__ char smem[];
  __bf16* xt   = (__bf16*)(smem + SM_XT);
  __bf16* pt   = (__bf16*)(smem + SM_PT);
  float*  red  = (float*)(smem + SM_RED);
  float*  mrun = (float*)(smem + SM_STAT);
  float*  lrun = mrun + M_BLK;
  float*  mnew = lrun + M_BLK;
  float*  linv = mnew + M_BLK;

  const int tid  = threadIdx.x;
  const int w    = tid >> 5;
  const int lane = tid & 31;
  const int half = lane >> 4;
  const int l16  = lane & 15;
  const int m0   = blockIdx.x * M_BLK;

  // ---- Stage x row block (32 x 1024 bf16, 64KB) into LDS via the Tensor
  // Data Mover (ISA ch.8: D# group0 = flags/lds_addr/global_addr, group1 =
  // dims/strides; groups 2-3 + trailing group zero for a 2D tensor). Wave 0
  // issues the DMA (TDM ignores EXEC; one wave avoids 32 duplicate
  // transfers), waits on TENSORcnt, then the barrier publishes the tile.
#if __has_builtin(__builtin_amdgcn_tensor_load_to_lds)
  if (tid < 32) {
    const unsigned long long ga =
        (unsigned long long)(const void*)(xb + (size_t)m0 * D_IN);
    v4u g0;
    g0[0] = 1u;                                    // count=1, user descriptor
    g0[1] = (unsigned)SM_XT;                       // lds_addr (bytes)
    g0[2] = (unsigned)(ga & 0xFFFFFFFFu);          // global_addr[31:0]
    g0[3] = (unsigned)((ga >> 32) & 0x01FFFFFFu)   // global_addr[56:32]
            | (2u << 30);                          // type = 2 ("image")
    v8i g1;
    g1[0] = 0x00010000;                            // data_size=1 (2 bytes)
    g1[1] = (int)((unsigned)(D_IN & 0xFFFF) << 16);    // tensor_dim0[15:0]
    g1[2] = (int)((unsigned)(NROWS & 0xFFFF) << 16);   // dim0[31:16]=0 | tensor_dim1[15:0]
    g1[3] = (int)((unsigned)(D_IN & 0xFFFF) << 16);    // dim1[31:16]=0 | tile_dim0
    g1[4] = M_BLK;                                 // tile_dim1 | tile_dim2=0
    g1[5] = D_IN;                                  // tensor_dim0_stride[31:0]
    g1[6] = 0;                                     // stride hi | dim1_stride lo
    g1[7] = 0;
    v4i gz4 = {0, 0, 0, 0};
    v8i gz8 = {0, 0, 0, 0, 0, 0, 0, 0};
    __builtin_amdgcn_tensor_load_to_lds(g0, g1, gz4, gz4, gz8, 0);
    __builtin_amdgcn_s_wait_tensorcnt(0);
  }
#else
  for (int i = tid; i < M_BLK * D_IN; i += NTHREADS)
    xt[i] = xb[(size_t)m0 * D_IN + i];
#endif
  if (tid < M_BLK) { mrun[tid] = -1e38f; lrun[tid] = 0.0f; }
  __syncthreads();

  const int NTILES = (VOCAB + VT - 1) / VT;   // 63 (last tile half-valid)

  // ---------- Pass 1: streaming row max / log-sum-exp over vocab ----------
  for (int t = 0; t < NTILES; ++t) {
    const int v0 = t * VT;
    v8f s0, s1;
    logits_tile(xt, Wb, bff, v0, w, lane, s0, s1);
    const bool valid = (v0 + w * 16 + l16) < VOCAB;

#pragma unroll
    for (int e = 0; e < 8; ++e) {
      float a = row16_max(valid ? s0[e] : -1e38f);
      float b = row16_max(valid ? s1[e] : -1e38f);
      if (l16 == 0) {
        red[w * M_BLK + e + 8 * half]      = a;
        red[w * M_BLK + 16 + e + 8 * half] = b;
      }
    }
    __syncthreads();
    if (tid < M_BLK) {
      float mx = -1e38f;
      for (int ww = 0; ww < NWAVES; ++ww) mx = fmaxf(mx, red[ww * M_BLK + tid]);
      mnew[tid] = fmaxf(mrun[tid], mx);
    }
    __syncthreads();
#pragma unroll
    for (int e = 0; e < 8; ++e) {
      const int r0 = e + 8 * half, r1 = 16 + r0;
      float a = row16_sum(valid ? __expf(s0[e] - mnew[r0]) : 0.0f);
      float b = row16_sum(valid ? __expf(s1[e] - mnew[r1]) : 0.0f);
      if (l16 == 0) { red[w * M_BLK + r0] = a; red[w * M_BLK + r1] = b; }
    }
    __syncthreads();
    if (tid < M_BLK) {
      float sm = 0.0f;
      for (int ww = 0; ww < NWAVES; ++ww) sm += red[ww * M_BLK + tid];
      const float alpha = __expf(mrun[tid] - mnew[tid]);
      lrun[tid] = lrun[tid] * alpha + sm;
      mrun[tid] = mnew[tid];
    }
    __syncthreads();
  }
  if (tid < M_BLK) linv[tid] = 1.0f / lrun[tid];
  __syncthreads();

  // ---------- Pass 2: probs (recomputed, final stats) @ emb ----------
  v8f acc0[8], acc1[8];
#pragma unroll
  for (int t = 0; t < 8; ++t)
#pragma unroll
    for (int e = 0; e < 8; ++e) { acc0[t][e] = 0.0f; acc1[t][e] = 0.0f; }

  for (int tv = 0; tv < NTILES; ++tv) {
    const int v0 = tv * VT;
    v8f s0, s1;
    logits_tile(xt, Wb, bff, v0, w, lane, s0, s1);
    const bool valid = (v0 + w * 16 + l16) < VOCAB;

#pragma unroll
    for (int e = 0; e < 8; ++e) {
      const int r0 = e + 8 * half, r1 = 16 + r0;
      const float p0 = valid ? __expf(s0[e] - mrun[r0]) * linv[r0] : 0.0f;
      const float p1 = valid ? __expf(s1[e] - mrun[r1]) * linv[r1] : 0.0f;
      pt[(size_t)r0 * VT + w * 16 + l16] = (__bf16)p0;
      pt[(size_t)r1 * VT + w * 16 + l16] = (__bf16)p1;
    }
    __syncthreads();

    // probs[32 x 512] @ emb[512 x 4096]; wave w owns D_OUT cols [w*128, w*128+128).
    // k0 outer so the two prob A-fragments are reused across all 8 n-tiles.
    const int kgp = ((v0 <= VOCAB - VT - 32) ? (v0 + VT) : (VOCAB - 32));
#pragma unroll
    for (int t = 0; t < 8; ++t)   // prefetch next tile's embT rows (hoisted)
      __builtin_prefetch(embT + (size_t)(w * 128 + t * 16 + l16) * VOCAB + kgp, 0, 1);
#pragma unroll 2
    for (int k0 = 0; k0 < VT; k0 += 32) {
      const int kb = k0 + half * 8;
      v16bf a0 = frag_bf16(pt + (size_t)l16 * VT + kb);
      v16bf a1 = frag_bf16(pt + (size_t)(16 + l16) * VT + kb);
      // clamp so the OOB vocab tail (probs == 0) never reads past embT
      int kg = v0 + kb; kg = (kg <= VOCAB - 32) ? kg : (VOCAB - 32);
#pragma unroll
      for (int t = 0; t < 8; ++t) {
        const size_t nrow = (size_t)(w * 128 + t * 16 + l16) * VOCAB;
        v16bf b = frag_bf16(embT + nrow + kg);
        acc0[t] = __builtin_amdgcn_wmma_f32_16x16x32_bf16(false, a0, false, b, (short)0, acc0[t], false, false);
        acc1[t] = __builtin_amdgcn_wmma_f32_16x16x32_bf16(false, a1, false, b, (short)0, acc1[t], false, false);
      }
    }
    __syncthreads();  // protect pt before next tile overwrites it
  }

  // ---------- Store 32 x 4096 f32 output ----------
#pragma unroll
  for (int t = 0; t < 8; ++t) {
    const int n = w * 128 + t * 16 + l16;
#pragma unroll
    for (int e = 0; e < 8; ++e) {
      const int r0 = e + 8 * half;
      out[(size_t)(m0 + r0) * D_OUT + n]      = acc0[t][e];
      out[(size_t)(m0 + 16 + r0) * D_OUT + n] = acc1[t][e];
    }
  }
}

extern "C" void kernel_launch(void* const* d_in, const int* in_sizes, int n_in,
                              void* d_out, int out_size, void* d_ws, size_t ws_size,
                              hipStream_t stream) {
  const float* x   = (const float*)d_in[0];
  const float* W   = (const float*)d_in[1];
  const float* bff = (const float*)d_in[2];
  const float* emb = (const float*)d_in[3];
  float* out = (float*)d_out;

  // Workspace: Wb [VOCAB x D_IN] bf16 (64MB), embT [D_OUT x VOCAB] bf16 (256MB),
  //            xb [NROWS x D_IN] bf16 (8MB)
  char* ws = (char*)d_ws;
  __bf16* Wb   = (__bf16*)ws;
  __bf16* embT = (__bf16*)(ws + (size_t)VOCAB * D_IN * 2);
  __bf16* xb   = (__bf16*)(ws + (size_t)VOCAB * D_IN * 2 + (size_t)D_OUT * VOCAB * 2);

  // Prep: f32 -> bf16 for W_ff and x; emb transposed to bf16.
  {
    int n = VOCAB * D_IN;
    cvt_f32_bf16<<<dim3(n / 8 / 256), dim3(256), 0, stream>>>(W, Wb, n);
  }
  {
    int n = NROWS * D_IN;
    cvt_f32_bf16<<<dim3(n / 8 / 256), dim3(256), 0, stream>>>(x, xb, n);
  }
  transpose_emb<<<dim3(VOCAB / TT, D_OUT / TT), dim3(256), 0, stream>>>(emb, embT);

  static_assert(SM_TOTAL <= 320 * 1024, "LDS budget");
  (void)hipFuncSetAttribute((const void*)fused_ff_softmax_emb,
                            hipFuncAttributeMaxDynamicSharedMemorySize, (int)SM_TOTAL);

  dim3 grid(NROWS / M_BLK);       // 128 workgroups
  dim3 block(NTHREADS);           // 32 waves (wave32)
  fused_ff_softmax_emb<<<grid, block, SM_TOTAL, stream>>>(xb, Wb, bff, embT, out);
}
